// PatchedSelfAttention_69741678952768
// MI455X (gfx1250) — compile-verified, matrix operand order checked by reading, not run on hardware
//
#include <hip/hip_runtime.h>

#define HDIM 4096
#define NHE  32
#define HD   128
#define KVH  2
#define GSZ  (NHE / KVH)
#define BATCH 4
#define SEQ  1024
#define TTOK (BATCH * SEQ)
#define QKVD (NHE * HD + 2 * KVH * HD) /* 4608 */

typedef __bf16 bf16;
typedef __attribute__((ext_vector_type(16))) __bf16 v16bf;
typedef __attribute__((ext_vector_type(8)))  float  v8f;
typedef int v4i_ __attribute__((vector_size(16)));

#define AS1 __attribute__((address_space(1)))
#define AS3 __attribute__((address_space(3)))

#if __has_builtin(__builtin_amdgcn_global_load_async_to_lds_b128)
#define USE_ASYNC_LDS 1
#else
#define USE_ASYNC_LDS 0
#endif

union FragU { v16bf v; unsigned u[8]; };

// 16x32 bf16 fragment load from a row-major tile (ld elements, even).
// ISA layout: lane m=lane&15, half h=lane>>4, VGPR v holds the contiguous
// pair K = 16*(v>>2) + 8*h + 2*(v&3) — folds into two ds_load_b128 per lane.
__device__ __forceinline__ v16bf load_frag(const bf16* base, int ld) {
  const int lane = threadIdx.x & 31;
  const int m = lane & 15;
  const int h = lane >> 4;
  FragU f;
  const bf16* row = base + m * ld;
#pragma unroll
  for (int v = 0; v < 8; ++v) {
    const int k0 = ((v >> 2) << 4) + (h << 3) + ((v & 3) << 1);
    f.u[v] = *(const unsigned*)(row + k0);
  }
  return f.v;
}

// Copy one 64-byte row (32 bf16) global -> LDS.
__device__ __forceinline__ void row_to_lds(const bf16* g, bf16* s) {
#if USE_ASYNC_LDS
  AS1 v4i_* gp = (AS1 v4i_*)(__UINTPTR_TYPE__)g;
  AS3 v4i_* sp = (AS3 v4i_*)(unsigned)(__UINTPTR_TYPE__)s;
#pragma unroll
  for (int q = 0; q < 4; ++q)
    __builtin_amdgcn_global_load_async_to_lds_b128(gp + q, sp + q, 0, 0);
#else
  const uint4* src = (const uint4*)g;
  uint4* dst = (uint4*)s;
  uint4 d0 = src[0], d1 = src[1], d2 = src[2], d3 = src[3];
  dst[0] = d0; dst[1] = d1; dst[2] = d2; dst[3] = d3;
#endif
}

__device__ __forceinline__ void async_fence() {
#if USE_ASYNC_LDS
#if __has_builtin(__builtin_amdgcn_s_wait_asynccnt)
  __builtin_amdgcn_s_wait_asynccnt(0);
#else
  asm volatile("s_wait_asynccnt 0x0" ::: "memory");
#endif
#endif
}

// ---------------- conversion kernels ----------------
__global__ void k_f32_to_bf16(const float* __restrict__ src, bf16* __restrict__ dst,
                              long long n) {
  long long i = (long long)blockIdx.x * blockDim.x + threadIdx.x;
  const long long stride = (long long)gridDim.x * blockDim.x;
  for (; i < n; i += stride) dst[i] = (bf16)src[i];
}

// W: K x N row-major fp32 -> Wt: N x K row-major bf16 (LDS-tiled, coalesced)
__global__ __launch_bounds__(256)
void k_transpose_bf16(const float* __restrict__ W, bf16* __restrict__ Wt,
                      int K, int N) {
  __shared__ float tile[32][33];
  const int bx = blockIdx.x * 32; // N
  const int by = blockIdx.y * 32; // K
  const int tx = threadIdx.x;     // 0..31
  const int ty = threadIdx.y;     // 0..7
#pragma unroll
  for (int i = 0; i < 32; i += 8)
    tile[ty + i][tx] = W[(size_t)(by + ty + i) * N + bx + tx];
  __syncthreads();
#pragma unroll
  for (int i = 0; i < 32; i += 8)
    Wt[(size_t)(bx + ty + i) * K + by + tx] = (bf16)tile[tx][ty + i];
}

// ------- bf16 WMMA GEMM: C[M,N] = A[M,K] * Bt[N,K]^T (+bias) --------------
// 128 threads = 4 waves, block tile 128x128, wave tile 64x64 (4x4 WMMA),
// double-buffered LDS, one barrier per 32-wide K step.
__global__ __launch_bounds__(128)
void k_gemm_bf16(const bf16* __restrict__ A, const bf16* __restrict__ Bt,
                 const float* __restrict__ bias, float* __restrict__ C,
                 int M, int N, int K) {
  __shared__ bf16 sA[2][128 * 32];
  __shared__ bf16 sB[2][128 * 32];
  const int tid  = threadIdx.x;
  const int row0 = blockIdx.y * 128;
  const int col0 = blockIdx.x * 128;
  const int wave = tid >> 5;
  const int wr   = (wave >> 1) << 6; // 0,64
  const int wc   = (wave & 1) << 6;  // 0,64
  v8f acc[4][4] = {};

  const bf16* gA = A  + (size_t)(row0 + tid) * K;
  const bf16* gB = Bt + (size_t)(col0 + tid) * K;

  row_to_lds(gA, &sA[0][tid * 32]);
  row_to_lds(gB, &sB[0][tid * 32]);

  const int NK = K >> 5;
  for (int ks = 0; ks < NK; ++ks) {
    async_fence();
    __syncthreads();
    if (ks + 1 < NK) {
      const int nb = (ks + 1) & 1;
      row_to_lds(gA + (ks + 1) * 32, &sA[nb][tid * 32]);
      row_to_lds(gB + (ks + 1) * 32, &sB[nb][tid * 32]);
    }
    const bf16* a0 = &sA[ks & 1][0];
    const bf16* b0 = &sB[ks & 1][0];
    v16bf af[4], bfr[4];
#pragma unroll
    for (int i = 0; i < 4; ++i) af[i]  = load_frag(a0 + (wr + 16 * i) * 32, 32);
#pragma unroll
    for (int j = 0; j < 4; ++j) bfr[j] = load_frag(b0 + (wc + 16 * j) * 32, 32);
#pragma unroll
    for (int i = 0; i < 4; ++i)
#pragma unroll
      for (int j = 0; j < 4; ++j)
        acc[i][j] = __builtin_amdgcn_wmma_f32_16x16x32_bf16(
            false, af[i], false, bfr[j], (short)0, acc[i][j], false, false);
  }

  const int lane = tid & 31;
  const int n  = lane & 15;
  const int hh = lane >> 4;
#pragma unroll
  for (int i = 0; i < 4; ++i)
#pragma unroll
    for (int j = 0; j < 4; ++j) {
      const int col = col0 + wc + 16 * j + n;
      const float bv = bias ? bias[col] : 0.0f;
#pragma unroll
      for (int v = 0; v < 8; ++v) {
        const int rr = row0 + wr + 16 * i + v + (hh << 3);
        C[(size_t)rr * N + col] = acc[i][j][v] + bv;
      }
    }
}

// ---------------- RoPE + split + bf16 convert ----------------
__global__ __launch_bounds__(64)
void k_rope_split(const float* __restrict__ mixed, const float* __restrict__ rope,
                  bf16* __restrict__ Qb, bf16* __restrict__ Kb, bf16* __restrict__ Vb) {
  const int t    = blockIdx.x;  // token
  const int slot = blockIdx.y;  // 0..35
  const int d    = threadIdx.x; // dims 2d, 2d+1
  const int b = t / SEQ, s = t % SEQ;
  const float* src;
  bf16* dst;
  bool dorope = true;
  if (slot < NHE) {
    src = mixed + (size_t)t * QKVD + slot * HD;
    dst = Qb + (((size_t)(b * NHE + slot)) * SEQ + s) * HD;
  } else if (slot < NHE + KVH) {
    const int kv = slot - NHE;
    src = mixed + (size_t)t * QKVD + NHE * HD + kv * HD;
    dst = Kb + (((size_t)(b * KVH + kv)) * SEQ + s) * HD;
  } else {
    const int kv = slot - NHE - KVH;
    src = mixed + (size_t)t * QKVD + (NHE + KVH) * HD + kv * HD;
    dst = Vb + (((size_t)(b * KVH + kv)) * SEQ + s) * HD;
    dorope = false;
  }
  const float x0 = src[2 * d], x1 = src[2 * d + 1];
  if (dorope && d < 32) {
    const float c  = rope[((size_t)s * 32 + d) * 2];
    const float sn = rope[((size_t)s * 32 + d) * 2 + 1];
    dst[2 * d]     = (bf16)(x0 * c - x1 * sn);
    dst[2 * d + 1] = (bf16)(x1 * c + x0 * sn);
  } else {
    dst[2 * d]     = (bf16)x0;
    dst[2 * d + 1] = (bf16)x1;
  }
}

// ---------------- flash attention (causal, GQA) ----------------
// grid: (SEQ/64, NHE, BATCH), block: 128 (4 waves x 16 q-rows)
__global__ __launch_bounds__(128)
void k_attn(const bf16* __restrict__ Qb, const bf16* __restrict__ Kb,
            const bf16* __restrict__ Vb, bf16* __restrict__ ctx) {
  __shared__ bf16 sK[32 * 128];   // keys x dims
  __shared__ bf16 sVt[128 * 32];  // dims x keys (transposed)
  __shared__ bf16 sP[4][16 * 32]; // per-wave P staging

  const int b  = blockIdx.z;
  const int qh = blockIdx.y;
  const int kv = qh / GSZ;
  const int row0 = blockIdx.x * 64;
  const int tid  = threadIdx.x;
  const int wave = tid >> 5;
  const int lane = tid & 31;
  const int q0   = row0 + wave * 16;

  const bf16* Qp = Qb + (((size_t)(b * NHE + qh)) * SEQ + q0) * HD;
  v16bf aq[4];
#pragma unroll
  for (int kk = 0; kk < 4; ++kk) aq[kk] = load_frag(Qp + kk * 32, HD);

  v8f acc[8] = {};
  float mrow[8], lrow[8];
#pragma unroll
  for (int v = 0; v < 8; ++v) { mrow[v] = -1e30f; lrow[v] = 0.0f; }

  const float scale = 0.088388347648318447f; // 1/sqrt(128)
  const bf16* Kbase = Kb + ((size_t)(b * KVH + kv)) * SEQ * HD;
  const bf16* Vbase = Vb + ((size_t)(b * KVH + kv)) * SEQ * HD;
  const int nsteps = (row0 + 64) / 32;
  const int n  = lane & 15;
  const int hh = lane >> 4;

  for (int step = 0; step < nsteps; ++step) {
    const int k0 = step * 32;
    {
      const int r   = tid >> 2;
      const int seg = (tid & 3) << 5;
      const uint4* gk = (const uint4*)(Kbase + (size_t)(k0 + r) * HD + seg);
      uint4* dk = (uint4*)(sK + r * HD + seg);
      dk[0] = gk[0]; dk[1] = gk[1]; dk[2] = gk[2]; dk[3] = gk[3];
      const bf16* gv = Vbase + (size_t)(k0 + r) * HD + seg;
#pragma unroll
      for (int dd = 0; dd < 32; ++dd) sVt[(seg + dd) * 32 + r] = gv[dd];
    }
    __syncthreads();

    v8f sc[2] = {};
#pragma unroll
    for (int c = 0; c < 2; ++c)
#pragma unroll
      for (int kk = 0; kk < 4; ++kk) {
        v16bf bfr = load_frag(sK + (c * 16) * HD + kk * 32, HD);
        sc[c] = __builtin_amdgcn_wmma_f32_16x16x32_bf16(
            false, aq[kk], false, bfr, (short)0, sc[c], false, false);
      }

#pragma unroll
    for (int c = 0; c < 2; ++c)
#pragma unroll
      for (int v = 0; v < 8; ++v) {
        const int qi = q0 + v + (hh << 3);
        const int ki = k0 + c * 16 + n;
        const float x = sc[c][v] * scale;
        sc[c][v] = (ki <= qi) ? x : -1e30f;
      }

#pragma unroll
    for (int v = 0; v < 8; ++v) {
      float rmax = fmaxf(sc[0][v], sc[1][v]);
#pragma unroll
      for (int off = 8; off >= 1; off >>= 1)
        rmax = fmaxf(rmax, __shfl_xor(rmax, off, 32));
      const float newm  = fmaxf(mrow[v], rmax);
      const float alpha = __expf(mrow[v] - newm);
      const float p0 = __expf(sc[0][v] - newm);
      const float p1 = __expf(sc[1][v] - newm);
      float rs = p0 + p1;
#pragma unroll
      for (int off = 8; off >= 1; off >>= 1) rs += __shfl_xor(rs, off, 32);
      lrow[v] = lrow[v] * alpha + rs;
      mrow[v] = newm;
#pragma unroll
      for (int c8 = 0; c8 < 8; ++c8) acc[c8][v] *= alpha;
      sc[0][v] = p0; sc[1][v] = p1;
    }

#pragma unroll
    for (int c = 0; c < 2; ++c)
#pragma unroll
      for (int v = 0; v < 8; ++v)
        sP[wave][(v + (hh << 3)) * 32 + c * 16 + n] = (bf16)sc[c][v];
    __syncthreads();

    v16bf pa = load_frag(&sP[wave][0], 32);
#pragma unroll
    for (int c8 = 0; c8 < 8; ++c8) {
      v16bf bv = load_frag(sVt + (c8 * 16) * 32, 32);
      acc[c8] = __builtin_amdgcn_wmma_f32_16x16x32_bf16(
          false, pa, false, bv, (short)0, acc[c8], false, false);
    }
    __syncthreads();
  }

  bf16* outp = ctx + ((size_t)(b * SEQ) + q0) * (NHE * HD) + qh * HD;
#pragma unroll
  for (int c8 = 0; c8 < 8; ++c8)
#pragma unroll
    for (int v = 0; v < 8; ++v) {
      const int rr = v + (hh << 3);
      outp[(size_t)rr * (NHE * HD) + c8 * 16 + n] = (bf16)(acc[c8][v] / lrow[v]);
    }
}

// ---------------- launch ----------------
extern "C" void kernel_launch(void* const* d_in, const int* in_sizes, int n_in,
                              void* d_out, int out_size, void* d_ws, size_t ws_size,
                              hipStream_t stream) {
  const float* hidden = (const float*)d_in[0];
  const float* rope   = (const float*)d_in[1];
  const float* Wqkv   = (const float*)d_in[2];
  const float* bqkv   = (const float*)d_in[3];
  const float* Wo     = (const float*)d_in[4];
  // d_in[5] (block_offsets): scatter+gather by same permutation == identity.

  char* ws = (char*)d_ws;
  size_t off = 0;
  auto carve = [&](size_t bytes) {
    void* p = ws + off;
    off = (off + bytes + 255) & ~(size_t)255;
    return p;
  };
  bf16*  Hb    = (bf16*)carve((size_t)TTOK * HDIM * 2);
  bf16*  WqkvT = (bf16*)carve((size_t)QKVD * HDIM * 2);
  float* mixed = (float*)carve((size_t)TTOK * QKVD * 4);
  bf16*  Kb    = (bf16*)carve((size_t)BATCH * KVH * SEQ * HD * 2);
  bf16*  Vb    = (bf16*)carve((size_t)BATCH * KVH * SEQ * HD * 2);
  bf16* Qb   = WqkvT;        // reuse after GEMM1
  bf16* ctxb = Hb;           // reuse after GEMM1
  bf16* WoT  = (bf16*)mixed; // reuse after rope

  k_f32_to_bf16<<<2048, 256, 0, stream>>>(hidden, Hb, (long long)TTOK * HDIM);
  k_transpose_bf16<<<dim3(QKVD / 32, HDIM / 32), dim3(32, 8), 0, stream>>>(
      Wqkv, WqkvT, HDIM, QKVD);

  dim3 g1(QKVD / 128, TTOK / 128);
  k_gemm_bf16<<<g1, 128, 0, stream>>>(Hb, WqkvT, bqkv, mixed, TTOK, QKVD, HDIM);

  dim3 g2(TTOK, NHE + 2 * KVH);
  k_rope_split<<<g2, 64, 0, stream>>>(mixed, rope, Qb, Kb, Vb);

  k_transpose_bf16<<<dim3(HDIM / 32, HDIM / 32), dim3(32, 8), 0, stream>>>(
      Wo, WoT, HDIM, HDIM);

  dim3 g3(SEQ / 64, NHE, BATCH);
  k_attn<<<g3, 128, 0, stream>>>(Qb, Kb, Vb, ctxb);

  dim3 g4(HDIM / 128, TTOK / 128);
  k_gemm_bf16<<<g4, 128, 0, stream>>>(ctxb, WoT, nullptr, (float*)d_out,
                                      TTOK, HDIM, HDIM);
}
